// MP_GCN_67448166417077
// MI455X (gfx1250) — matrix operand order repaired and checked by reading, not compiled, not run on hardware
//
#include <hip/hip_runtime.h>
#include <math.h>

// ---------------------------------------------------------------------------
// MP-GCN (attentional aggregation + GRUCell, 4 iters) for MI455X / gfx1250.
//
// Design rationale (no runtime available -> spec reasoning):
//  * Total GEMM FLOPs ~45 GF; edge gather/scatter ~0.9 GB/iter, but the whole
//    node state (h 25.6MB + macc 25.6MB) fits in the 192MB L2, so the random
//    per-edge traffic is L2-resident.  The kernel set is therefore
//    scatter/atomic bound, not HBM or compute bound.
//  * GEMMs (GRU: [N,128]x[128,384] x2 per iter; final attention GEMMs) use
//    v_wmma_f32_16x16x32_bf16: bf16 operands, f32 accumulate (K<=256).
//  * Edge kernels: one wave32 per edge; each lane moves a float4 of the
//    512B feature row (fully coalesced), scatter via global_atomic_add_f32.
//  * Segment-max via monotonic-uint float encoding + atomicMax; the self-loop
//    seeds it so every segment is non-empty (matches reference semantics).
//
// Workspace use: ~53 MB (h, macc, per-node scalars, bf16 weights, pooled).
// ---------------------------------------------------------------------------

typedef __attribute__((ext_vector_type(16))) __bf16 v16bf;
typedef __attribute__((ext_vector_type(8)))  float  v8f;

#define FDIM 128     // feature width
#define G3   384     // 3*F (GRU gate width)
#define H1   85      // int(128/1.5)
#define H2   64
#define PROP_ITER 4

// ---------------- small device helpers ----------------
__device__ __forceinline__ unsigned short f2bf(float f) {
  unsigned u = __float_as_uint(f);
  unsigned r = u + 0x7FFFu + ((u >> 16) & 1u);   // round-to-nearest-even
  return (unsigned short)(r >> 16);
}
__device__ __forceinline__ float ssign(float v) { return v / (1.0f + fabsf(v)); }
__device__ __forceinline__ float sigm(float v)  { return 1.0f / (1.0f + expf(-v)); }

// monotonic float -> uint key (for atomicMax-based segment max)
__device__ __forceinline__ unsigned fkey(float f) {
  unsigned u = __float_as_uint(f);
  return (u & 0x80000000u) ? ~u : (u | 0x80000000u);
}
__device__ __forceinline__ float fkey_inv(unsigned k) {
  unsigned u = (k & 0x80000000u) ? (k & 0x7FFFFFFFu) : ~k;
  return __uint_as_float(u);
}

// One 16x16 (M x N) f32 tile of A[16,K] x B[K,N-tile], bf16 operands.
// aP/bP are u32 views of bf16 buffers (2 elems per u32).
//  A layout (16-bit, 16x32 per k-step): lane L -> M = L&15; half = L>>4
//    VGPR q<4 : K = 32*kb + half*8  + 2q{+1};  q>=4 : K += 16
//  B layout (16-bit, 32x16): lane L -> N = L&15;  same K mapping as A
//  D layout: VGPR v -> row v + 8*half, col = L&15
__device__ __forceinline__ v8f bf16_tile_mm(const unsigned* aP, int aStrideU32, int aOffU32,
                                            const unsigned* bP, int bStrideU32, int j0,
                                            int nkb, int lane, v8f acc) {
  const int half = lane >> 4;
  const int mrow = lane & 15;
  for (int kb = 0; kb < nkb; ++kb) {
    union { v16bf v; unsigned u[8]; } A, B;
    const int ab = mrow * aStrideU32 + aOffU32 + kb * 16 + half * 4;
#pragma unroll
    for (int q = 0; q < 4; ++q) { A.u[q] = aP[ab + q]; A.u[q + 4] = aP[ab + 8 + q]; }
    const int bb = (j0 + mrow) * bStrideU32 + kb * 16 + half * 4;
#pragma unroll
    for (int q = 0; q < 4; ++q) { B.u[q] = bP[bb + q]; B.u[q + 4] = bP[bb + 8 + q]; }
    acc = __builtin_amdgcn_wmma_f32_16x16x32_bf16(false, A.v, false, B.v,
                                                  (short)0, acc, false, false);
  }
  return acc;
}

// ---------------- prep kernels ----------------
__global__ void k_copy_f32(const float* __restrict__ src, float* __restrict__ dst, int n) {
  for (int i = blockIdx.x * blockDim.x + threadIdx.x; i < n; i += gridDim.x * blockDim.x)
    dst[i] = src[i];
}
__global__ void k_conv_bf(const float* __restrict__ src, unsigned short* __restrict__ dst, int n) {
  for (int i = blockIdx.x * blockDim.x + threadIdx.x; i < n; i += gridDim.x * blockDim.x)
    dst[i] = f2bf(src[i]);
}
// src is [K,J] row-major; dst is bf16 [J,K] (transposed) so B-tile reads are
// contiguous along K.
__global__ void k_conv_bf_T(const float* __restrict__ src, unsigned short* __restrict__ dst,
                            int K, int J) {
  for (int i = blockIdx.x * blockDim.x + threadIdx.x; i < K * J; i += gridDim.x * blockDim.x) {
    int k = i / J, j = i % J;
    dst[j * K + k] = f2bf(src[i]);
  }
}
__global__ void k_zero_f32(float* __restrict__ p, int n) {
  for (int i = blockIdx.x * blockDim.x + threadIdx.x; i < n; i += gridDim.x * blockDim.x)
    p[i] = 0.0f;
}

// ---------------- gate MLP: g[node] (128 -> 64 -> 32 -> 1) ----------------
// one 64-thread block per node (tiny: ~1 GFLOP total)
__global__ void k_gate(const float* __restrict__ h,
                       const float* __restrict__ W1, const float* __restrict__ b1,
                       const float* __restrict__ W2, const float* __restrict__ b2,
                       const float* __restrict__ W3, const float* __restrict__ b3,
                       float* __restrict__ g) {
  __shared__ float sh[FDIM];
  __shared__ float o1[64];
  __shared__ float o2[32];
  const int node = blockIdx.x;
  const int t = threadIdx.x;
  sh[t]      = h[(size_t)node * FDIM + t];
  sh[t + 64] = h[(size_t)node * FDIM + t + 64];
  __syncthreads();
  float a = b1[t];
  for (int k = 0; k < FDIM; ++k) a += sh[k] * W1[k * 64 + t];
  o1[t] = ssign(a);
  __syncthreads();
  if (t < 32) {
    float c = b2[t];
    for (int k = 0; k < 64; ++k) c += o1[k] * W2[k * 32 + t];
    o2[t] = ssign(c);
  }
  __syncthreads();
  if (t == 0) {
    float s = b3[0];
    for (int k = 0; k < 32; ++k) s += o2[k] * W3[k];
    g[node] = s;
  }
}

// ---------------- segment softmax over edges ----------------
// seed segment-max with the self-loop logit
__global__ void k_init_mx(const float* __restrict__ g, unsigned* __restrict__ mxkey, int n) {
  for (int i = blockIdx.x * blockDim.x + threadIdx.x; i < n; i += gridDim.x * blockDim.x)
    mxkey[i] = fkey(g[i]);
}
// one thread per directed edge instance (symmetrized: 2E)
__global__ void k_edge_max(const int* __restrict__ ei, const float* __restrict__ eattr,
                           const float* __restrict__ g, unsigned* __restrict__ mxkey,
                           const float* __restrict__ tcut, int E) {
  const float tc = tcut[0];
  for (int i = blockIdx.x * blockDim.x + threadIdx.x; i < 2 * E; i += gridDim.x * blockDim.x) {
    const int e = (i < E) ? i : i - E;
    int s, d;
    if (i < E) { s = ei[e]; d = ei[E + e]; } else { s = ei[E + e]; d = ei[e]; }
    if (eattr[e] <= tc) atomicMax(&mxkey[d], fkey(g[s]));
  }
}
// self-loop contribution: mx decode, denom = exp(g-mx), macc = w_self * h
__global__ void k_self(const float* __restrict__ g, const unsigned* __restrict__ mxkey,
                       const float* __restrict__ h, float* __restrict__ mxf,
                       float* __restrict__ denom, float* __restrict__ macc, int n) {
  for (int i = blockIdx.x * blockDim.x + threadIdx.x; i < n * FDIM; i += gridDim.x * blockDim.x) {
    const int node = i >> 7;
    const float mv = fkey_inv(mxkey[node]);
    const float ws = expf(g[node] - mv);
    if ((i & 127) == 0) { mxf[node] = mv; denom[node] = ws; }
    macc[i] = ws * h[i];
  }
}
// one wave32 per edge: coalesced 512B row gather + f32 atomic scatter
__global__ void k_edge_acc(const int* __restrict__ ei, const float* __restrict__ eattr,
                           const float* __restrict__ g, const float* __restrict__ mxf,
                           const float* __restrict__ h, float* __restrict__ denom,
                           float* __restrict__ macc, const float* __restrict__ tcut, int E) {
  const int gtid = blockIdx.x * blockDim.x + threadIdx.x;
  const int wid  = gtid >> 5;
  const int lane = gtid & 31;
  if (wid >= 2 * E) return;
  const int e = (wid < E) ? wid : wid - E;
  int s, d;
  if (wid < E) { s = ei[e]; d = ei[E + e]; } else { s = ei[E + e]; d = ei[e]; }
  if (eattr[e] > tcut[0]) return;                     // edge masked out
  const float w = expf(g[s] - mxf[d]);
  if (lane == 0) atomicAdd(&denom[d], w);
  const float4 hv = ((const float4*)(h + (size_t)s * FDIM))[lane];
  float* mp = macc + (size_t)d * FDIM + lane * 4;
  atomicAdd(mp + 0, w * hv.x);
  atomicAdd(mp + 1, w * hv.y);
  atomicAdd(mp + 2, w * hv.z);
  atomicAdd(mp + 3, w * hv.w);
}

// ---------------- GRUCell via WMMA ----------------
// 16 nodes per block (N % 16 == 0), 256 threads = 8 waves.
// gi = (macc/denom) @ Wih^T ; gh = h @ Whh^T ; then fused gate math.
__global__ void k_gru(const float* __restrict__ macc, const float* __restrict__ denom,
                      float* __restrict__ h,
                      const unsigned short* __restrict__ WihB,
                      const unsigned short* __restrict__ WhhB,
                      const float* __restrict__ bih, const float* __restrict__ bhh) {
  __shared__ __align__(16) unsigned short sAm[16 * FDIM];  // bf16 normalized m
  __shared__ __align__(16) unsigned short sAh[16 * FDIM];  // bf16 h
  __shared__ float sGi[16 * G3];
  __shared__ float sGh[16 * G3];

  const int row0 = blockIdx.x * 16;
  const int tid  = threadIdx.x;
  if (tid == 0) { __builtin_prefetch(WihB, 0, 3); __builtin_prefetch(WhhB, 0, 3); }

  for (int i = tid; i < 16 * FDIM; i += 256) {
    const int r = i >> 7, k = i & 127;
    const int node = row0 + r;
    const float mval = macc[(size_t)node * FDIM + k] / denom[node];
    sAm[i] = f2bf(mval);
    sAh[i] = f2bf(h[(size_t)node * FDIM + k]);
  }
  __syncthreads();

  const int wave = tid >> 5;
  const int lane = tid & 31;
  const int half = lane >> 4;
  const int mrow = lane & 15;
  const unsigned* aM = (const unsigned*)sAm;
  const unsigned* aH = (const unsigned*)sAh;
  const unsigned* bI = (const unsigned*)WihB;   // [384][128] bf16, K contiguous
  const unsigned* bH = (const unsigned*)WhhB;

  for (int t = 0; t < 3; ++t) {
    const int j0 = wave * 48 + t * 16;
    v8f z = {0.f,0.f,0.f,0.f,0.f,0.f,0.f,0.f};
    v8f gi = bf16_tile_mm(aM, 64, 0, bI, 64, j0, 4, lane, z);
    v8f gh = bf16_tile_mm(aH, 64, 0, bH, 64, j0, 4, lane, z);
#pragma unroll
    for (int v = 0; v < 8; ++v) {
      const int r = v + 8 * half;
      sGi[r * G3 + j0 + mrow] = gi[v];
      sGh[r * G3 + j0 + mrow] = gh[v];
    }
  }
  __syncthreads();

  for (int i = tid; i < 16 * FDIM; i += 256) {
    const int r = i >> 7, j = i & 127;
    const int node = row0 + r;
    const float rr   = sigm(sGi[r * G3 + j]       + bih[j]       + sGh[r * G3 + j]       + bhh[j]);
    const float zz   = sigm(sGi[r * G3 + 128 + j] + bih[128 + j] + sGh[r * G3 + 128 + j] + bhh[128 + j]);
    const float cand = tanhf(sGi[r * G3 + 256 + j] + bih[256 + j] +
                             rr * (sGh[r * G3 + 256 + j] + bhh[256 + j]));
    const float hold = h[(size_t)node * FDIM + j];
    h[(size_t)node * FDIM + j] = (1.0f - zz) * cand + zz * hold;
  }
}

// ---------------- final attention + pooling via WMMA ----------------
// ai = ssign(ssign([h|x]@W1+b1)@W2+b2); attn = softmax(ai, feat);
// aj = ssign(x@Wj+bj); pooled[batch] += attn*aj
__global__ void k_final(const float* __restrict__ h, const float* __restrict__ x,
                        const unsigned short* __restrict__ W1t,  // bf16 [128][256]
                        const float* __restrict__ b1,
                        const unsigned short* __restrict__ W2t,  // bf16 [128][128]
                        const float* __restrict__ b2,
                        const unsigned short* __restrict__ Wjt,  // bf16 [128][128]
                        const float* __restrict__ bj,
                        const int* __restrict__ batch, float* __restrict__ pooled) {
  __shared__ __align__(16) unsigned short sA[16 * 256];    // [h|x] bf16
  __shared__ __align__(16) unsigned short sAi[16 * FDIM];  // layer-1 out, bf16
  __shared__ float sAtt[16 * FDIM];
  __shared__ float sAj[16 * FDIM];

  const int row0 = blockIdx.x * 16;
  const int tid  = threadIdx.x;
  for (int i = tid; i < 16 * 256; i += 256) {
    const int r = i >> 8, c = i & 255;
    const int node = row0 + r;
    const float v = (c < FDIM) ? h[(size_t)node * FDIM + c]
                               : x[(size_t)node * FDIM + (c - FDIM)];
    sA[i] = f2bf(v);
  }
  __syncthreads();

  const int wave = tid >> 5;
  const int lane = tid & 31;
  const int half = lane >> 4;
  const int mrow = lane & 15;
  const int j0 = wave * 16;
  const unsigned* aP  = (const unsigned*)sA;
  const unsigned* aiP = (const unsigned*)sAi;
  const v8f zero = {0.f,0.f,0.f,0.f,0.f,0.f,0.f,0.f};

  // layer 1: [h|x] (K=256) @ W1 -> softsign -> bf16 LDS
  v8f c1 = bf16_tile_mm(aP, 128, 0, (const unsigned*)W1t, 128, j0, 8, lane, zero);
  const int col = j0 + mrow;
#pragma unroll
  for (int v = 0; v < 8; ++v) {
    const int r = v + 8 * half;
    sAi[r * FDIM + col] = f2bf(ssign(c1[v] + b1[col]));
  }
  __syncthreads();

  // layer 2 (attention logits) and aj branch (independent)
  v8f c2 = bf16_tile_mm(aiP, 64, 0,  (const unsigned*)W2t, 64, j0, 4, lane, zero);
  v8f c3 = bf16_tile_mm(aP, 128, 64, (const unsigned*)Wjt, 64, j0, 4, lane, zero); // x part
#pragma unroll
  for (int v = 0; v < 8; ++v) {
    const int r = v + 8 * half;
    sAtt[r * FDIM + col] = ssign(c2[v] + b2[col]);
    sAj[r * FDIM + col]  = ssign(c3[v] + bj[col]);
  }
  __syncthreads();

  // per-row softmax over features, multiply, pool by graph
  if (tid < 16) {
    const int r = tid;
    const int node = row0 + r;
    float mx = -1e30f;
    for (int j = 0; j < FDIM; ++j) mx = fmaxf(mx, sAtt[r * FDIM + j]);
    float s = 0.0f;
    for (int j = 0; j < FDIM; ++j) {
      const float e = expf(sAtt[r * FDIM + j] - mx);
      sAtt[r * FDIM + j] = e;
      s += e;
    }
    const float inv = 1.0f / s;
    const int b = batch[node];
    for (int j = 0; j < FDIM; ++j)
      atomicAdd(&pooled[(size_t)b * FDIM + j], sAtt[r * FDIM + j] * inv * sAj[r * FDIM + j]);
  }
}

// ---------------- output MLP: 128 -> 85 -> 64 -> 1 per graph ----------------
__global__ void k_out(const float* __restrict__ pooled,
                      const float* __restrict__ W1, const float* __restrict__ b1,
                      const float* __restrict__ W2, const float* __restrict__ b2,
                      const float* __restrict__ W3, const float* __restrict__ b3,
                      float* __restrict__ out) {
  __shared__ float sp[FDIM];
  __shared__ float o1[H1];
  __shared__ float o2[H2];
  const int gidx = blockIdx.x;
  const int t = threadIdx.x;
  sp[t] = pooled[(size_t)gidx * FDIM + t];
  __syncthreads();
  if (t < H1) {
    float a = b1[t];
    for (int k = 0; k < FDIM; ++k) a += sp[k] * W1[k * H1 + t];
    o1[t] = fmaxf(a, 0.0f);
  }
  __syncthreads();
  if (t < H2) {
    float a = b2[t];
    for (int k = 0; k < H1; ++k) a += o1[k] * W2[k * H2 + t];
    o2[t] = fmaxf(a, 0.0f);
  }
  __syncthreads();
  if (t == 0) {
    float a = b3[0];
    for (int k = 0; k < H2; ++k) a += o2[k] * W3[k];
    out[gidx] = a;
  }
}

// ---------------------------------------------------------------------------
static inline int cdiv(int a, int b) { return (a + b - 1) / b; }

extern "C" void kernel_launch(void* const* d_in, const int* in_sizes, int n_in,
                              void* d_out, int out_size, void* d_ws, size_t ws_size,
                              hipStream_t stream) {
  const float* x     = (const float*)d_in[0];
  const int*   ei    = (const int*)d_in[1];     // [2,E] flat: row0=src, row1=dst
  const float* eattr = (const float*)d_in[2];
  const int*   batch = (const int*)d_in[3];
  const float* tcut  = (const float*)d_in[4];
  const float* gW1 = (const float*)d_in[5],  *gb1 = (const float*)d_in[6];
  const float* gW2 = (const float*)d_in[7],  *gb2 = (const float*)d_in[8];
  const float* gW3 = (const float*)d_in[9],  *gb3 = (const float*)d_in[10];
  const float* Wih = (const float*)d_in[11], *Whh = (const float*)d_in[12];
  const float* bih = (const float*)d_in[13], *bhh = (const float*)d_in[14];
  const float* aiW1 = (const float*)d_in[15], *aib1 = (const float*)d_in[16];
  const float* aiW2 = (const float*)d_in[17], *aib2 = (const float*)d_in[18];
  const float* ajW  = (const float*)d_in[19], *ajb  = (const float*)d_in[20];
  const float* oW1 = (const float*)d_in[21], *ob1 = (const float*)d_in[22];
  const float* oW2 = (const float*)d_in[23], *ob2 = (const float*)d_in[24];
  const float* oW3 = (const float*)d_in[25], *ob3 = (const float*)d_in[26];

  const int N  = in_sizes[0] / FDIM;   // 50000 (divisible by 16)
  const int E  = in_sizes[2];          // 400000
  const int NG = out_size;             // 50 graphs (OUT==1)

  // workspace carve-out (~53 MB), 256B aligned slices
  size_t off = 0;
  auto alloc = [&](size_t bytes) -> void* {
    void* p = (char*)d_ws + off;
    off += (bytes + 255) & ~(size_t)255;
    return p;
  };
  float*          h      = (float*)alloc((size_t)N * FDIM * 4);
  float*          macc   = (float*)alloc((size_t)N * FDIM * 4);
  float*          g      = (float*)alloc((size_t)N * 4);
  unsigned*       mxkey  = (unsigned*)alloc((size_t)N * 4);
  float*          mxf    = (float*)alloc((size_t)N * 4);
  float*          denom  = (float*)alloc((size_t)N * 4);
  unsigned short* WihB   = (unsigned short*)alloc((size_t)G3 * FDIM * 2);
  unsigned short* WhhB   = (unsigned short*)alloc((size_t)G3 * FDIM * 2);
  unsigned short* W1tB   = (unsigned short*)alloc((size_t)128 * 256 * 2);
  unsigned short* W2tB   = (unsigned short*)alloc((size_t)128 * 128 * 2);
  unsigned short* WjtB   = (unsigned short*)alloc((size_t)128 * 128 * 2);
  float*          pooled = (float*)alloc((size_t)NG * FDIM * 4);
  (void)ws_size; (void)n_in;

  // prep: h = x ; bf16 weight conversions (GRU weights kept [j][k]; attention
  // weights transposed to [j][k] so WMMA B-tiles read contiguous K)
  k_copy_f32 <<<cdiv(N * FDIM, 256), 256, 0, stream>>>(x, h, N * FDIM);
  k_conv_bf  <<<cdiv(G3 * FDIM, 256), 256, 0, stream>>>(Wih, WihB, G3 * FDIM);
  k_conv_bf  <<<cdiv(G3 * FDIM, 256), 256, 0, stream>>>(Whh, WhhB, G3 * FDIM);
  k_conv_bf_T<<<cdiv(256 * 128, 256), 256, 0, stream>>>(aiW1, W1tB, 256, 128);
  k_conv_bf_T<<<cdiv(128 * 128, 256), 256, 0, stream>>>(aiW2, W2tB, 128, 128);
  k_conv_bf_T<<<cdiv(128 * 128, 256), 256, 0, stream>>>(ajW,  WjtB, 128, 128);

  for (int it = 0; it < PROP_ITER; ++it) {
    k_gate    <<<N, 64, 0, stream>>>(h, gW1, gb1, gW2, gb2, gW3, gb3, g);
    k_init_mx <<<cdiv(N, 256), 256, 0, stream>>>(g, mxkey, N);
    k_edge_max<<<cdiv(2 * E, 256), 256, 0, stream>>>(ei, eattr, g, mxkey, tcut, E);
    k_self    <<<cdiv(N * FDIM, 256), 256, 0, stream>>>(g, mxkey, h, mxf, denom, macc, N);
    k_edge_acc<<<cdiv(2 * E, 8), 256, 0, stream>>>(ei, eattr, g, mxf, h, denom, macc, tcut, E);
    k_gru     <<<N / 16, 256, 0, stream>>>(macc, denom, h, WihB, WhhB, bih, bhh);
  }

  k_zero_f32<<<cdiv(NG * FDIM, 256), 256, 0, stream>>>(pooled, NG * FDIM);
  k_final   <<<N / 16, 256, 0, stream>>>(h, x, W1tB, aib1, W2tB, aib2, WjtB, ajb,
                                         batch, pooled);
  k_out     <<<NG, FDIM, 0, stream>>>(pooled, oW1, ob1, oW2, ob2, oW3, ob3,
                                      (float*)d_out);
}
